// AtomicSidechainEncoder_76192719831702
// MI455X (gfx1250) — matrix operand order repaired and checked by reading, not compiled, not run on hardware
//
#include <hip/hip_runtime.h>

// ---------------------------------------------------------------------------
// AtomicSidechainEncoder for MI455X (gfx1250, wave32, WMMA).
// v2: LDS tiles stored pre-swizzled in the WMMA wave32 register layout as f16
// (1 x 32B load per fragment instead of 16 scalar f32 loads + cvts), and all
// divisions replaced by v_rcp_f32.
// ---------------------------------------------------------------------------

#define NUM_RES   2048
#define APR       14
#define NUM_ATOMS (NUM_RES * APR)   // 28672
#define E_R       786432
#define NLAYERS   4

typedef __attribute__((ext_vector_type(16))) _Float16 v16h;
typedef __attribute__((ext_vector_type(8)))  float    v8f;

#define WMMA_F32F16(a, b, c) \
  __builtin_amdgcn_wmma_f32_16x16x32_f16(false, (a), false, (b), (short)0, (c), false, false)

// ---------------------------------------------------------------------------
// Swizzled f16 fragment tiles: one tile = 512 halves (1KB), lane-major
// [32 lanes][16 halves].  Layout math per ISA 7.12.2 (wave32).
// ---------------------------------------------------------------------------
__device__ __forceinline__ unsigned packh2(float a, float b) {
  union { _Float16 h[2]; unsigned u; } un;
  un.h[0] = (_Float16)a;
  un.h[1] = (_Float16)b;
  return un.u;
}

__device__ __forceinline__ v16h ldfrag(const _Float16* fb, int lane) {
  return *(const v16h*)(fb + lane * 16);
}

// A-matrix (16xK per 32-K step): element (m, k) -> lane, half
__device__ __forceinline__ int a_lane(int m, int k) { return m + 16 * ((k >> 3) & 1); }
__device__ __forceinline__ int a_half(int k)        { return (k & 7) + 8 * (k >> 4); }
// B-matrix (32x16 tile): element (k, n) -> lane, half
__device__ __forceinline__ int b_lane(int n, int k) { return n + 16 * ((k >> 3) & 1); }
__device__ __forceinline__ int b_half(int k)        { return (k & 7) + 8 * (k >> 4); }

__device__ __forceinline__ void stA_pair(_Float16* fb, int m, int k, float v0, float v1) {
  *(unsigned*)(fb + a_lane(m, k) * 16 + a_half(k)) = packh2(v0, v1);   // k even
}
__device__ __forceinline__ void stA_one(_Float16* fb, int m, int k, float v) {
  fb[a_lane(m, k) * 16 + a_half(k)] = (_Float16)v;
}
__device__ __forceinline__ void stB_pair(_Float16* fb, int n, int k, float v0, float v1) {
  *(unsigned*)(fb + b_lane(n, k) * 16 + b_half(k)) = packh2(v0, v1);   // k even
}

// Stage a row-major KxN f32 weight (K mult of 32, N mult of 16) into swizzled
// f16 B-fragment tiles; tile index = ks*(N/16) + nt.
__device__ __forceinline__ void stageW(_Float16* f, const float* W, int K, int N,
                                       int t, int nth) {
  int npairs = (K >> 1) * N;
  for (int i = t; i < npairs; i += nth) {
    int kp = i / N, n = i - kp * N;
    int k = kp << 1;
    int kl = k & 31;
    _Float16* tile = f + (((k >> 5) * (N >> 4)) + (n >> 4)) * 512;
    stB_pair(tile, n & 15, kl, W[k * N + n], W[(k + 1) * N + n]);
  }
}

__device__ __forceinline__ float frcp(float x)     { return __builtin_amdgcn_rcpf(x); }
__device__ __forceinline__ float sigm(float v)     { return frcp(1.f + __expf(-v)); }
__device__ __forceinline__ float siluf(float v)    { return v * sigm(v); }

__device__ __forceinline__ void atomicMaxFloat(float* addr, float value) {
  if (value >= 0.f) atomicMax((int*)addr, __float_as_int(value));
  else              atomicMin((unsigned int*)addr, __float_as_uint(value));
}

// ---------------------------------------------------------------------------
// Elementwise / setup kernels
// ---------------------------------------------------------------------------
__global__ void k_fill(float* p, float v, int n) {
  int i = blockIdx.x * blockDim.x + threadIdx.x;
  if (i < n) p[i] = v;
}

__global__ void k_copy(float* dst, const float* src, int n) {
  int i = blockIdx.x * blockDim.x + threadIdx.x;
  if (i < n) dst[i] = src[i];
}

__global__ void k_embed(const float* __restrict__ emb, const float* __restrict__ w,
                        const float* __restrict__ b, float* __restrict__ x) {
  int i = blockIdx.x * blockDim.x + threadIdx.x;
  if (i >= NUM_ATOMS * 128) return;
  int a = i >> 7, j = i & 127;
  if (j < 32) {
    float s = b[j];
    const float* er = emb + (size_t)a * 23;
#pragma unroll
    for (int k = 0; k < 23; ++k) s += er[k] * w[k * 32 + j];
    x[i] = s;
  } else {
    x[i] = 0.f;
  }
}

// Edge rotation matrices + (optional) RBF features. R rows = [y; n; z].
__global__ void k_geom(const float* __restrict__ pos, const int* __restrict__ src,
                       const int* __restrict__ dst, int E, float* __restrict__ R,
                       float* __restrict__ ef) {
  int e = blockIdx.x * blockDim.x + threadIdx.x;
  if (e >= E) return;
  int s = src[e], d0 = dst[e];
  float vx = pos[(size_t)d0 * 3 + 0] - pos[(size_t)s * 3 + 0];
  float vy = pos[(size_t)d0 * 3 + 1] - pos[(size_t)s * 3 + 1];
  float vz = pos[(size_t)d0 * 3 + 2] - pos[(size_t)s * 3 + 2];
  float dd = sqrtf(vx * vx + vy * vy + vz * vz);
  float inv = frcp(fmaxf(dd, 1e-8f));
  float nx = vx * inv, ny = vy * inv, nz = vz * inv;
  float ux, uy, uz;
  if (fabsf(nz) < 0.9f) { ux = 0.f; uy = 0.f; uz = 1.f; }
  else                  { ux = 1.f; uy = 0.f; uz = 0.f; }
  float zx = ny * uz - nz * uy, zy = nz * ux - nx * uz, zz = nx * uy - ny * ux;
  float zn = frcp(fmaxf(sqrtf(zx * zx + zy * zy + zz * zz), 1e-8f));
  zx *= zn; zy *= zn; zz *= zn;
  float yx = zy * nz - zz * ny, yy = zz * nx - zx * nz, yz = zx * ny - zy * nx;
  size_t b = (size_t)e * 9;
  R[b + 0] = yx; R[b + 1] = yy; R[b + 2] = yz;
  R[b + 3] = nx; R[b + 4] = ny; R[b + 5] = nz;
  R[b + 6] = zx; R[b + 7] = zy; R[b + 8] = zz;
  if (ef) {
#pragma unroll
    for (int k = 0; k < 32; ++k) {
      float mu = 20.f * (float)k / 31.f;
      float t = (dd - mu) * 1.6f;   // sigma = 0.625
      ef[(size_t)e * 32 + k] = __expf(-t * t);
    }
  }
}

// ---------------------------------------------------------------------------
// trans_block stage 1: attention logits.  WPB=4 (block=128), 16 edges / wave.
// ---------------------------------------------------------------------------
__global__ void k_attn(const float* __restrict__ x, const float* __restrict__ ef,
                       const int* __restrict__ src, const int* __restrict__ dst, int E,
                       const float* __restrict__ aw1, const float* __restrict__ ab1,
                       const float* __restrict__ aw2, float* __restrict__ logits) {
  const int WPB = 4;
  __shared__ __align__(32) _Float16 sW1f[3 * 512];     // 96x16
  __shared__ __align__(32) _Float16 sW2f[512];         // aw2 zero-padded 32x16
  __shared__ __align__(32) _Float16 sInvF[WPB][3 * 512];
  __shared__ __align__(32) _Float16 sHF[WPB][512];
  int t = threadIdx.x;
  __builtin_prefetch(aw1, 0, 3);
  stageW(sW1f, aw1, 96, 16, t, 128);
  for (int i = t; i < 256; i += 128) {
    int kp = i >> 4, n = i & 15, k = kp << 1;
    float v0 = (k < 16 && n < 8) ? aw2[k * 8 + n] : 0.f;
    float v1 = (k + 1 < 16 && n < 8) ? aw2[(k + 1) * 8 + n] : 0.f;
    stB_pair(sW2f, n, k, v0, v1);
  }
  __syncthreads();
  int wave = t >> 5, lane = t & 31;
  int e0 = (blockIdx.x * WPB + wave) * 16;
  _Float16* invF = sInvF[wave];
  _Float16* hF = sHF[wave];
  {
    int i = lane >> 1, cb = (lane & 1) * 16;
    int e = e0 + i;
    int s = src[e], d0 = dst[e];
    const float* xs = x + (size_t)s * 128;
    const float* xd = x + (size_t)d0 * 128;
    const float* fe = ef + (size_t)e * 32;
    for (int c = 0; c < 16; c += 2) {
      stA_pair(invF,        i, cb + c, xs[cb + c], xs[cb + c + 1]);
      stA_pair(invF + 512,  i, cb + c, xd[cb + c], xd[cb + c + 1]);
      stA_pair(invF + 1024, i, cb + c, fe[cb + c], fe[cb + c + 1]);
    }
  }
  __syncthreads();
  int n = lane & 15, hi = lane >> 4;
  v8f acc = {};
#pragma unroll
  for (int ks = 0; ks < 3; ++ks) {
    v16h a = ldfrag(invF + ks * 512, lane);
    v16h b = ldfrag(sW1f + ks * 512, lane);
    acc = WMMA_F32F16(a, b, acc);
  }
#pragma unroll
  for (int r = 0; r < 8; ++r) {
    int m = r + 8 * hi;
    float v = siluf(acc[r] + ab1[n]);
    stA_one(hF, m, n, v);
    stA_one(hF, m, 16 + n, 0.f);
  }
  __syncthreads();
  v8f acc2 = {};
  {
    v16h a = ldfrag(hF, lane);
    v16h b = ldfrag(sW2f, lane);
    acc2 = WMMA_F32F16(a, b, acc2);
  }
  if (n < 8) {
#pragma unroll
    for (int r = 0; r < 8; ++r) {
      int m = r + 8 * hi;
      logits[(size_t)(e0 + m) * 8 + n] = acc2[r];
    }
  }
}

// ---------------------------------------------------------------------------
// segment softmax (3 atomic passes over E*8)
// ---------------------------------------------------------------------------
__global__ void k_segmax(const float* __restrict__ lg, const int* __restrict__ dst,
                         int E, float* __restrict__ segmax) {
  int i = blockIdx.x * blockDim.x + threadIdx.x;
  if (i >= E * 8) return;
  int e = i >> 3, h = i & 7;
  atomicMaxFloat(&segmax[(size_t)dst[e] * 8 + h], lg[i]);
}

__global__ void k_segexp(float* __restrict__ lg, const int* __restrict__ dst, int E,
                         const float* __restrict__ segmax, float* __restrict__ segsum) {
  int i = blockIdx.x * blockDim.x + threadIdx.x;
  if (i >= E * 8) return;
  int e = i >> 3, h = i & 7;
  float m = segmax[(size_t)dst[e] * 8 + h];
  if (!(m >= -3.0e38f)) m = 0.f;  // isfinite fallback
  float v = __expf(lg[i] - m);
  lg[i] = v;
  atomicAdd(&segsum[(size_t)dst[e] * 8 + h], v);
}

__global__ void k_segnorm(float* __restrict__ attn, const int* __restrict__ dst, int E,
                          const float* __restrict__ segsum) {
  int i = blockIdx.x * blockDim.x + threadIdx.x;
  if (i >= E * 8) return;
  int e = i >> 3, h = i & 7;
  attn[i] = attn[i] * frcp(segsum[(size_t)dst[e] * 8 + h] + 1e-9f);
}

// ---------------------------------------------------------------------------
// trans_block stage 2: messages.  WPB=2 (block=64), 16 edges / wave.
// ---------------------------------------------------------------------------
__global__ void k_message(const float* __restrict__ x, const float* __restrict__ Rm,
                          const float* __restrict__ attn, const int* __restrict__ src,
                          const int* __restrict__ dst, int E, const float* __restrict__ vw,
                          const float* __restrict__ ow, float* __restrict__ acc) {
  const int WPB = 2;
  __shared__ __align__(32) _Float16 sVWf[4 * 512];   // 32x64
  __shared__ __align__(32) _Float16 sOWf[4 * 512];   // 64x32
  __shared__ __align__(32) _Float16 sXSf[WPB][4 * 512];
  __shared__ __align__(32) _Float16 sVTf[WPB][2 * 512];
  __shared__ float sM[WPB][512];
  __shared__ float sR[WPB][144];
  __shared__ float sA[WPB][128];
  int t = threadIdx.x;
  __builtin_prefetch(vw, 0, 3);
  __builtin_prefetch(ow, 0, 3);
  stageW(sVWf, vw, 32, 64, t, 64);
  stageW(sOWf, ow, 64, 32, t, 64);
  int wave = t >> 5, lane = t & 31;
  int e0 = (blockIdx.x * WPB + wave) * 16;
  _Float16* xsF = sXSf[wave];
  _Float16* vtF = sVTf[wave];
  float* mtile = sM[wave];
  float* Rw = sR[wave];
  float* at = sA[wave];
  for (int i = lane; i < 144; i += 32) Rw[i] = Rm[(size_t)e0 * 9 + i];
  for (int i = lane; i < 128; i += 32) at[i] = attn[(size_t)e0 * 8 + i];
  __syncthreads();
  {  // stage xs = rot(x[src]) directly in fragment layout
    int i = lane >> 1, cb = (lane & 1) * 16;
    int s = src[e0 + i];
    const float* xr = x + (size_t)s * 128;
    const float* Re = Rw + i * 9;
    for (int c = 0; c < 16; c += 2) {
      int k = cb + c;
      float a0 = xr[k],      b0 = xr[k + 1];
      float a1 = xr[32 + k], b1 = xr[33 + k];
      float a2 = xr[64 + k], b2 = xr[65 + k];
      float a3 = xr[96 + k], b3 = xr[97 + k];
#pragma unroll
      for (int l = 0; l < 4; ++l) {
        float v0, v1;
        if (l == 0) { v0 = a0; v1 = b0; }
        else {
          const float* rr = Re + (l - 1) * 3;
          v0 = rr[0] * a1 + rr[1] * a2 + rr[2] * a3;
          v1 = rr[0] * b1 + rr[1] * b2 + rr[2] * b3;
        }
        int row = i * 4 + l;
        stA_pair(xsF + (row >> 4) * 512, row & 15, k, v0, v1);
      }
    }
  }
  __syncthreads();
  int n = lane & 15, hi = lane >> 4;
  v16h bv0 = ldfrag(sVWf, lane),        bv1 = ldfrag(sVWf + 512, lane);
  v16h bv2 = ldfrag(sVWf + 1024, lane), bv3 = ldfrag(sVWf + 1536, lane);
  v16h bo00 = ldfrag(sOWf, lane),        bo01 = ldfrag(sOWf + 512, lane);
  v16h bo10 = ldfrag(sOWf + 1024, lane), bo11 = ldfrag(sOWf + 1536, lane);
  for (int mt = 0; mt < 4; ++mt) {
    v16h a = ldfrag(xsF + mt * 512, lane);
#pragma unroll
    for (int nt = 0; nt < 4; ++nt) {
      v16h b = (nt == 0) ? bv0 : (nt == 1) ? bv1 : (nt == 2) ? bv2 : bv3;
      v8f c = {};
      c = WMMA_F32F16(a, b, c);
#pragma unroll
      for (int r = 0; r < 8; ++r) {
        int m = r + 8 * hi;
        int ei = (mt * 16 + m) >> 2;          // local edge 0..15
        int col = nt * 16 + n;
        float v = c[r] * at[ei * 8 + (col >> 3)];
        stA_one(vtF + (col >> 5) * 512, m, col & 31, v);
      }
    }
    __syncthreads();
    v16h a20 = ldfrag(vtF, lane);
    v16h a21 = ldfrag(vtF + 512, lane);
#pragma unroll
    for (int nt = 0; nt < 2; ++nt) {
      v8f c = {};
      c = WMMA_F32F16(a20, (nt == 0) ? bo00 : bo01, c);
      c = WMMA_F32F16(a21, (nt == 0) ? bo10 : bo11, c);
#pragma unroll
      for (int r = 0; r < 8; ++r) mtile[(r + 8 * hi) * 32 + nt * 16 + n] = c[r];
    }
    __syncthreads();
    {  // rot_t + scatter (4 edges in this M-tile)
      int ri = lane >> 1, cb = (lane & 1) * 16;
      int eloc = mt * 4 + (ri >> 2);
      int l = ri & 3;
      int d0 = dst[e0 + eloc];
      const float* Re = Rw + eloc * 9;
      float* out = acc + (size_t)d0 * 128;
      if (l == 0) {
        for (int c = 0; c < 16; ++c) atomicAdd(&out[cb + c], mtile[ri * 32 + cb + c]);
      } else {
        int rb = ri - l;
        int i0 = l - 1;
        for (int c = 0; c < 16; ++c) {
          float m1 = mtile[(rb + 1) * 32 + cb + c];
          float m2 = mtile[(rb + 2) * 32 + cb + c];
          float m3 = mtile[(rb + 3) * 32 + cb + c];
          atomicAdd(&out[l * 32 + cb + c], Re[i0] * m1 + Re[3 + i0] * m2 + Re[6 + i0] * m3);
        }
      }
    }
    __syncthreads();
  }
}

// ---------------------------------------------------------------------------
// trans_block stage 3: x += acc, gated feed-forward. WPB=4 (block=128).
// ---------------------------------------------------------------------------
__global__ void k_nodeff(float* __restrict__ x, const float* __restrict__ acc,
                         const float* __restrict__ fw, const float* __restrict__ fb, int N) {
  const int WPB = 4;
  __shared__ __align__(32) _Float16 sWf[6 * 512];   // fw0 | fw1 | fw2, 2 tiles each
  __shared__ float sX0[WPB][512];
  __shared__ __align__(32) _Float16 sX0f[WPB][512];
  __shared__ __align__(32) _Float16 sSf[WPB][512];
  int t = threadIdx.x;
  __builtin_prefetch(fw, 0, 3);
  stageW(sWf,        fw,        32, 32, t, 128);
  stageW(sWf + 1024, fw + 1024, 32, 32, t, 128);
  stageW(sWf + 2048, fw + 2048, 32, 32, t, 128);
  int wave = t >> 5, lane = t & 31;
  int n0 = (blockIdx.x * WPB + wave) * 16;
  float* x0 = sX0[wave];
  _Float16* x0F = sX0f[wave];
  _Float16* ssF = sSf[wave];
  {
    int i = lane >> 1, cb = (lane & 1) * 16;
    size_t base = (size_t)(n0 + i) * 128;
    for (int c = 0; c < 16; c += 2) {
      float v0 = x[base + cb + c] + acc[base + cb + c];
      float v1 = x[base + cb + c + 1] + acc[base + cb + c + 1];
      x0[i * 32 + cb + c] = v0;
      x0[i * 32 + cb + c + 1] = v1;
      stA_pair(x0F, i, cb + c, v0, v1);
    }
  }
  __syncthreads();
  int n = lane & 15, hi = lane >> 4;
  {
    v16h a = ldfrag(x0F, lane);
#pragma unroll
    for (int nt = 0; nt < 2; ++nt) {
      v16h b = ldfrag(sWf + nt * 512, lane);
      v8f c = {};
      c = WMMA_F32F16(a, b, c);
#pragma unroll
      for (int r = 0; r < 8; ++r) {
        int m = r + 8 * hi;
        stA_one(ssF, m, nt * 16 + n, siluf(c[r] + fb[nt * 16 + n]));
      }
    }
  }
  __syncthreads();
  {
    v16h a = ldfrag(ssF, lane);
#pragma unroll
    for (int nt = 0; nt < 2; ++nt) {
      v16h b1f = ldfrag(sWf + (2 + nt) * 512, lane);
      v16h b2f = ldfrag(sWf + (4 + nt) * 512, lane);
      v8f c1 = {}; c1 = WMMA_F32F16(a, b1f, c1);
      v8f c2 = {}; c2 = WMMA_F32F16(a, b2f, c2);
#pragma unroll
      for (int r = 0; r < 8; ++r) {
        int m = r + 8 * hi;
        int col = nt * 16 + n;
        size_t base = (size_t)(n0 + m) * 128;
        float g = c1[r] + fb[32 + col];
        float gate = sigm(c2[r]);
        x[base + col] = x0[m * 32 + col] + g;
#pragma unroll
        for (int l = 1; l < 4; ++l) {
          float xl = x[base + l * 32 + col] + acc[base + l * 32 + col];
          x[base + l * 32 + col] = xl * (1.f + gate);
        }
      }
    }
  }
}

// ---------------------------------------------------------------------------
// edge_update: ef = ef + silu(inv@w1 + b1)@w2 + b2.  WPB=4 (block=128).
// ---------------------------------------------------------------------------
__global__ void k_edgeupd(const float* __restrict__ x, float* __restrict__ ef,
                          const int* __restrict__ src, const int* __restrict__ dst, int E,
                          const float* __restrict__ w1, const float* __restrict__ b1,
                          const float* __restrict__ w2, const float* __restrict__ b2) {
  const int WPB = 4;
  __shared__ __align__(32) _Float16 sW1f[6 * 512];   // 96x32: (ks*2+nt)
  __shared__ __align__(32) _Float16 sW2f[2 * 512];   // 32x32
  __shared__ __align__(32) _Float16 sInvF[WPB][3 * 512];
  __shared__ float sEfc[WPB][512];
  __shared__ __align__(32) _Float16 sHF[WPB][512];
  int t = threadIdx.x;
  __builtin_prefetch(w1, 0, 3);
  stageW(sW1f, w1, 96, 32, t, 128);
  stageW(sW2f, w2, 32, 32, t, 128);
  int wave = t >> 5, lane = t & 31;
  int e0 = (blockIdx.x * WPB + wave) * 16;
  _Float16* invF = sInvF[wave];
  float* efc = sEfc[wave];
  _Float16* hF = sHF[wave];
  {
    int i = lane >> 1, cb = (lane & 1) * 16;
    int e = e0 + i;
    int s = src[e], d0 = dst[e];
    const float* xs = x + (size_t)s * 128;
    const float* xd = x + (size_t)d0 * 128;
    const float* fe = ef + (size_t)e * 32;
    for (int c = 0; c < 16; c += 2) {
      int k = cb + c;
      stA_pair(invF,        i, k, xs[k], xs[k + 1]);
      stA_pair(invF + 512,  i, k, xd[k], xd[k + 1]);
      float e0v = fe[k], e1v = fe[k + 1];
      stA_pair(invF + 1024, i, k, e0v, e1v);
      efc[i * 32 + k] = e0v;
      efc[i * 32 + k + 1] = e1v;
    }
  }
  __syncthreads();
  int n = lane & 15, hi = lane >> 4;
  v16h a0 = ldfrag(invF, lane);
  v16h a1 = ldfrag(invF + 512, lane);
  v16h a2 = ldfrag(invF + 1024, lane);
#pragma unroll
  for (int nt = 0; nt < 2; ++nt) {
    v8f c = {};
    c = WMMA_F32F16(a0, ldfrag(sW1f + (0 * 2 + nt) * 512, lane), c);
    c = WMMA_F32F16(a1, ldfrag(sW1f + (1 * 2 + nt) * 512, lane), c);
    c = WMMA_F32F16(a2, ldfrag(sW1f + (2 * 2 + nt) * 512, lane), c);
#pragma unroll
    for (int r = 0; r < 8; ++r) {
      int m = r + 8 * hi;
      stA_one(hF, m, nt * 16 + n, siluf(c[r] + b1[nt * 16 + n]));
    }
  }
  __syncthreads();
  v16h a = ldfrag(hF, lane);
#pragma unroll
  for (int nt = 0; nt < 2; ++nt) {
    v16h b = ldfrag(sW2f + nt * 512, lane);
    v8f c = {};
    c = WMMA_F32F16(a, b, c);
#pragma unroll
    for (int r = 0; r < 8; ++r) {
      int m = r + 8 * hi;
      int col = nt * 16 + n;
      ef[(size_t)(e0 + m) * 32 + col] = efc[m * 32 + col] + c[r] + b2[col];
    }
  }
}

// ---------------------------------------------------------------------------
// residue gather / projection kernels
// ---------------------------------------------------------------------------
__global__ void k_gather_res(const float* __restrict__ x, const int* __restrict__ bbsel,
                             float* __restrict__ res) {
  int i = blockIdx.x * blockDim.x + threadIdx.x;
  if (i >= NUM_RES * 128) return;
  int r = i >> 7, j = i & 127;
  res[i] = x[(size_t)bbsel[r * 4 + 1] * 128 + j];
}

__global__ void k_gather_cax(const float* __restrict__ ax, const int* __restrict__ bbsel,
                             float* __restrict__ cax) {
  int i = blockIdx.x * blockDim.x + threadIdx.x;
  if (i >= NUM_RES * 3) return;
  int r = i / 3, j = i - r * 3;
  cax[i] = ax[(size_t)bbsel[r * 4 + 1] * 3 + j];
}

// msg = rot_t((rot(res[src]) @ pj_cw) * rad) -> atomicAdd into resacc. WPB=2.
__global__ void k_proj_edge(const float* __restrict__ res, const float* __restrict__ Rr,
                            const float* __restrict__ efa, const int* __restrict__ rsrc,
                            const int* __restrict__ rdst, const float* __restrict__ prw,
                            const float* __restrict__ prb, const float* __restrict__ pcw,
                            float* __restrict__ resacc) {
  const int WPB = 2;
  __shared__ __align__(32) _Float16 sPRWf[2 * 512];
  __shared__ __align__(32) _Float16 sPCWf[2 * 512];
  __shared__ __align__(32) _Float16 sEFf[WPB][512];
  __shared__ float sRad[WPB][512];
  __shared__ __align__(32) _Float16 sXSf[WPB][4 * 512];
  __shared__ float sT[WPB][512];
  __shared__ float sR[WPB][144];
  int t = threadIdx.x;
  stageW(sPRWf, prw, 32, 32, t, 64);
  stageW(sPCWf, pcw, 32, 32, t, 64);
  int wave = t >> 5, lane = t & 31;
  int e0 = (blockIdx.x * WPB + wave) * 16;
  _Float16* efF = sEFf[wave];
  float* rad = sRad[wave];
  _Float16* xsF = sXSf[wave];
  float* tt = sT[wave];
  float* Rw = sR[wave];
  for (int i = lane; i < 144; i += 32) Rw[i] = Rr[(size_t)e0 * 9 + i];
  for (int i = lane; i < 256; i += 32) {
    int row = i >> 4, k = (i & 15) << 1;
    const float* fe = efa + (size_t)(e0 + row) * 32;
    stA_pair(efF, row, k, fe[k], fe[k + 1]);
  }
  __syncthreads();
  int n = lane & 15, hi = lane >> 4;
  {  // rad = silu(ef @ prw + prb)
    v16h a = ldfrag(efF, lane);
#pragma unroll
    for (int nt = 0; nt < 2; ++nt) {
      v16h b = ldfrag(sPRWf + nt * 512, lane);
      v8f c = {};
      c = WMMA_F32F16(a, b, c);
#pragma unroll
      for (int r = 0; r < 8; ++r) {
        int m = r + 8 * hi;
        rad[m * 32 + nt * 16 + n] = siluf(c[r] + prb[nt * 16 + n]);
      }
    }
  }
  {  // xs = rot(res[rsrc]) in fragment layout
    int i = lane >> 1, cb = (lane & 1) * 16;
    int s = rsrc[e0 + i];
    const float* xr = res + (size_t)s * 128;
    const float* Re = Rw + i * 9;
    for (int c = 0; c < 16; c += 2) {
      int k = cb + c;
      float a0 = xr[k],      b0 = xr[k + 1];
      float a1 = xr[32 + k], b1 = xr[33 + k];
      float a2 = xr[64 + k], b2 = xr[65 + k];
      float a3 = xr[96 + k], b3 = xr[97 + k];
#pragma unroll
      for (int l = 0; l < 4; ++l) {
        float v0, v1;
        if (l == 0) { v0 = a0; v1 = b0; }
        else {
          const float* rr = Re + (l - 1) * 3;
          v0 = rr[0] * a1 + rr[1] * a2 + rr[2] * a3;
          v1 = rr[0] * b1 + rr[1] * b2 + rr[2] * b3;
        }
        int row = i * 4 + l;
        stA_pair(xsF + (row >> 4) * 512, row & 15, k, v0, v1);
      }
    }
  }
  __syncthreads();
  v16h bc0 = ldfrag(sPCWf, lane);
  v16h bc1 = ldfrag(sPCWf + 512, lane);
  for (int mt = 0; mt < 4; ++mt) {
    v16h a = ldfrag(xsF + mt * 512, lane);
#pragma unroll
    for (int nt = 0; nt < 2; ++nt) {
      v8f c = {};
      c = WMMA_F32F16(a, (nt == 0) ? bc0 : bc1, c);
#pragma unroll
      for (int r = 0; r < 8; ++r) {
        int m = r + 8 * hi;
        int ei = (mt * 16 + m) >> 2;
        int col = nt * 16 + n;
        tt[m * 32 + col] = c[r] * rad[ei * 32 + col];
      }
    }
    __syncthreads();
    {
      int ri = lane >> 1, cb = (lane & 1) * 16;
      int eloc = mt * 4 + (ri >> 2);
      int l = ri & 3;
      int d0 = rdst[e0 + eloc];
      const float* Re = Rw + eloc * 9;
      float* out = resacc + (size_t)d0 * 128;
      if (l == 0) {
        for (int c = 0; c < 16; ++c) atomicAdd(&out[cb + c], tt[ri * 32 + cb + c]);
      } else {
        int rb = ri - l, i0 = l - 1;
        for (int c = 0; c < 16; ++c) {
          float m1 = tt[(rb + 1) * 32 + cb + c];
          float m2 = tt[(rb + 2) * 32 + cb + c];
          float m3 = tt[(rb + 3) * 32 + cb + c];
          atomicAdd(&out[l * 32 + cb + c], Re[i0] * m1 + Re[3 + i0] * m2 + Re[6 + i0] * m3);
        }
      }
    }
    __syncthreads();
  }
}

// out = resacc + res @ pj_sw   (per component row). WPB=2.
__global__ void k_proj_node(const float* __restrict__ resin, const float* __restrict__ resacc,
                            const float* __restrict__ psw, float* __restrict__ out) {
  const int WPB = 2;
  __shared__ __align__(32) _Float16 sWf[2 * 512];
  __shared__ __align__(32) _Float16 sRowsF[WPB][4 * 512];
  int t = threadIdx.x;
  stageW(sWf, psw, 32, 32, t, 64);
  int wave = t >> 5, lane = t & 31;
  int n0 = (blockIdx.x * WPB + wave) * 16;
  _Float16* rowsF = sRowsF[wave];
  for (int i = lane; i < 1024; i += 32) {
    int row = i >> 4, k = (i & 15) << 1;
    const float* src = resin + (size_t)n0 * 128 + (size_t)row * 32;
    stA_pair(rowsF + (row >> 4) * 512, row & 15, k, src[k], src[k + 1]);
  }
  __syncthreads();
  int n = lane & 15, hi = lane >> 4;
  v16h b0 = ldfrag(sWf, lane);
  v16h b1 = ldfrag(sWf + 512, lane);
  for (int mt = 0; mt < 4; ++mt) {
    v16h a = ldfrag(rowsF + mt * 512, lane);
#pragma unroll
    for (int nt = 0; nt < 2; ++nt) {
      v8f c = {};
      c = WMMA_F32F16(a, (nt == 0) ? b0 : b1, c);
#pragma unroll
      for (int r = 0; r < 8; ++r) {
        int m = r + 8 * hi;
        size_t idx = (size_t)n0 * 128 + (size_t)(mt * 16 + m) * 32 + nt * 16 + n;
        out[idx] = resacc[idx] + c[r];
      }
    }
  }
}

// ---------------------------------------------------------------------------
// Host-side orchestration
// ---------------------------------------------------------------------------
static inline int div_up(int a, int b) { return (a + b - 1) / b; }

static void run_softmax(float* attn, const int* dst, int E, int N, float* segmax,
                        float* segsum, hipStream_t stream) {
  int nh = N * 8;
  k_fill<<<div_up(nh, 256), 256, 0, stream>>>(segmax, -__builtin_huge_valf(), nh);
  k_fill<<<div_up(nh, 256), 256, 0, stream>>>(segsum, 0.f, nh);
  k_segmax<<<div_up(E * 8, 256), 256, 0, stream>>>(attn, dst, E, segmax);
  k_segexp<<<div_up(E * 8, 256), 256, 0, stream>>>(attn, dst, E, segmax, segsum);
  k_segnorm<<<div_up(E * 8, 256), 256, 0, stream>>>(attn, dst, E, segsum);
}

static void run_trans(float* x, const float* ef, const int* src, const int* dst, int E, int N,
                      const float* aw1, const float* ab1, const float* aw2, const float* vw,
                      const float* ow, const float* fw, const float* fb, const float* R,
                      float* attn, float* acc, float* segmax, float* segsum,
                      hipStream_t stream) {
  k_attn<<<E / 64, 128, 0, stream>>>(x, ef, src, dst, E, aw1, ab1, aw2, attn);
  run_softmax(attn, dst, E, N, segmax, segsum, stream);
  k_fill<<<div_up(N * 128, 256), 256, 0, stream>>>(acc, 0.f, N * 128);
  k_message<<<E / 32, 64, 0, stream>>>(x, R, attn, src, dst, E, vw, ow, acc);
  k_nodeff<<<N / 64, 128, 0, stream>>>(x, acc, fw, fb, N);
}

extern "C" void kernel_launch(void* const* d_in, const int* in_sizes, int n_in,
                              void* d_out, int out_size, void* d_ws, size_t ws_size,
                              hipStream_t stream) {
  (void)in_sizes; (void)n_in; (void)out_size; (void)ws_size;
  // ---- inputs (setup_inputs dict order) ----
  const float* atom_embedding = (const float*)d_in[0];
  const float* atom_x         = (const float*)d_in[1];
  const int*   redge          = (const int*)d_in[2];
  const int*   aedge          = (const int*)d_in[3];
  const int*   bbsel          = (const int*)d_in[4];
  const int*   resedge        = (const int*)d_in[5];
  const float* emb_w0 = (const float*)d_in[6];
  const float* emb_b0 = (const float*)d_in[7];
  const float* ly_aw1 = (const float*)d_in[8];
  const float* ly_ab1 = (const float*)d_in[9];
  const float* ly_aw2 = (const float*)d_in[10];
  const float* ly_vw  = (const float*)d_in[11];
  const float* ly_ow  = (const float*)d_in[12];
  const float* ly_fw  = (const float*)d_in[13];
  const float* ly_fb  = (const float*)d_in[14];
  const float* ag_aw1 = (const float*)d_in[15];
  const float* ag_ab1 = (const float*)d_in[16];
  const float* ag_aw2 = (const float*)d_in[17];
  const float* ag_vw  = (const float*)d_in[18];
  const float* ag_ow  = (const float*)d_in[19];
  const float* ag_fw  = (const float*)d_in[20];
  const float* ag_fb  = (const float*)d_in[21];
  const float* mu_aw1 = (const float*)d_in[22];
  const float* mu_ab1 = (const float*)d_in[23];
  const float* mu_aw2 = (const float*)d_in[24];
  const float* mu_vw  = (const float*)d_in[25];
  const float* mu_ow  = (const float*)d_in[26];
  const float* mu_fw  = (const float*)d_in[27];
  const float* mu_fb  = (const float*)d_in[28];
  const float* lv_aw1 = (const float*)d_in[29];
  const float* lv_ab1 = (const float*)d_in[30];
  const float* lv_aw2 = (const float*)d_in[31];
  const float* lv_vw  = (const float*)d_in[32];
  const float* lv_ow  = (const float*)d_in[33];
  const float* lv_fw  = (const float*)d_in[34];
  const float* lv_fb  = (const float*)d_in[35];
  const float* eu_w1  = (const float*)d_in[36];
  const float* eu_b1  = (const float*)d_in[37];
  const float* eu_w2  = (const float*)d_in[38];
  const float* eu_b2  = (const float*)d_in[39];
  const float* pj_rw  = (const float*)d_in[40];
  const float* pj_rb  = (const float*)d_in[41];
  const float* pj_cw  = (const float*)d_in[42];
  const float* pj_sw  = (const float*)d_in[43];

  // ---- workspace carve-up (floats, ~190 MB) ----
  float* ws     = (float*)d_ws;
  float* x      = ws;                                  // NUM_ATOMS*128
  float* R      = x + (size_t)NUM_ATOMS * 128;         // E_R*9
  float* ef     = R + (size_t)E_R * 9;                 // E_R*32
  float* attn   = ef + (size_t)E_R * 32;               // E_R*8
  float* acc    = attn + (size_t)E_R * 8;              // NUM_ATOMS*128
  float* segmax = acc + (size_t)NUM_ATOMS * 128;       // NUM_ATOMS*8
  float* segsum = segmax + (size_t)NUM_ATOMS * 8;      // NUM_ATOMS*8
  float* res    = segsum + (size_t)NUM_ATOMS * 8;      // NUM_RES*128
  float* resacc = res + (size_t)NUM_RES * 128;         // NUM_RES*128
  float* cax    = resacc + (size_t)NUM_RES * 128;      // NUM_RES*3
  float* Rr     = cax + (size_t)NUM_RES * 3;           // NUM_ATOMS*9

  const int* src = redge;
  const int* dst = redge + E_R;

  // ---- atom embedding + radius-edge geometry ----
  k_embed<<<NUM_ATOMS * 128 / 256, 256, 0, stream>>>(atom_embedding, emb_w0, emb_b0, x);
  k_geom<<<div_up(E_R, 256), 256, 0, stream>>>(atom_x, src, dst, E_R, R, ef);

  // ---- 4 transformer layers + edge updates ----
  for (int i = 0; i < NLAYERS; ++i) {
    run_trans(x, ef, src, dst, E_R, NUM_ATOMS,
              ly_aw1 + (size_t)i * 96 * 16, ly_ab1 + (size_t)i * 16,
              ly_aw2 + (size_t)i * 16 * 8, ly_vw + (size_t)i * 2048,
              ly_ow + (size_t)i * 2048, ly_fw + (size_t)i * 3072,
              ly_fb + (size_t)i * 64, R, attn, acc, segmax, segsum, stream);
    k_edgeupd<<<E_R / 64, 128, 0, stream>>>(x, ef, src, dst, E_R,
                                            eu_w1 + (size_t)i * 96 * 32,
                                            eu_b1 + (size_t)i * 32,
                                            eu_w2 + (size_t)i * 1024,
                                            eu_b2 + (size_t)i * 32);
  }

  // ---- aggregation block (reuse R/ef buffers as Ra/efa) ----
  const int* asrc = aedge;
  const int* adst = aedge + NUM_ATOMS;
  k_geom<<<div_up(NUM_ATOMS, 256), 256, 0, stream>>>(atom_x, asrc, adst, NUM_ATOMS, R, ef);
  run_trans(x, ef, asrc, adst, NUM_ATOMS, NUM_ATOMS,
            ag_aw1, ag_ab1, ag_aw2, ag_vw, ag_ow, ag_fw, ag_fb,
            R, attn, acc, segmax, segsum, stream);

  // ---- residue gather + res-edge geometry (R only; keep efa intact) ----
  k_gather_res<<<div_up(NUM_RES * 128, 256), 256, 0, stream>>>(x, bbsel, res);
  k_gather_cax<<<div_up(NUM_RES * 3, 256), 256, 0, stream>>>(atom_x, bbsel, cax);
  const int* rsrc = resedge;
  const int* rdst = resedge + NUM_ATOMS;
  k_geom<<<div_up(NUM_ATOMS, 256), 256, 0, stream>>>(cax, rsrc, rdst, NUM_ATOMS, Rr,
                                                     (float*)nullptr);

  // ---- projection ----
  k_fill<<<div_up(NUM_RES * 128, 256), 256, 0, stream>>>(resacc, 0.f, NUM_RES * 128);
  k_proj_edge<<<NUM_ATOMS / 32, 64, 0, stream>>>(res, Rr, ef, rsrc, rdst, pj_rw, pj_rb,
                                                 pj_cw, resacc);
  float* out_mu = (float*)d_out;
  float* out_lv = out_mu + (size_t)NUM_RES * 128;
  k_proj_node<<<NUM_RES / 32, 64, 0, stream>>>(res, resacc, pj_sw, out_mu);
  k_copy<<<div_up(NUM_RES * 128, 256), 256, 0, stream>>>(out_lv, out_mu, NUM_RES * 128);

  // ---- mu / lv heads (in-place on d_out halves) ----
  run_trans(out_mu, ef, rsrc, rdst, NUM_ATOMS, NUM_RES,
            mu_aw1, mu_ab1, mu_aw2, mu_vw, mu_ow, mu_fw, mu_fb,
            Rr, attn, acc, segmax, segsum, stream);
  run_trans(out_lv, ef, rsrc, rdst, NUM_ATOMS, NUM_RES,
            lv_aw1, lv_ab1, lv_aw2, lv_vw, lv_ow, lv_fw, lv_fb,
            Rr, attn, acc, segmax, segsum, stream);
}
